// TreeRNN_45887430590706
// MI455X (gfx1250) — compile-verified, matrix operand order checked by reading, not run on hardware
//
#include <hip/hip_runtime.h>
#include <hip/hip_bf16.h>
#include <math.h>

typedef __attribute__((ext_vector_type(16))) __bf16 v16bf;
typedef __attribute__((ext_vector_type(8)))  float  v8f;

#define NHID   256
#define BATCH  512
#define LEAVES 128
#define KDIM   512      // 2*NHID
#define NDIM   1280     // 5*NHID
#define NTILES 80       // NDIM/16
#define NT_PER_WAVE 10
#define THREADS 256
#define LDSA_STRIDE 520   // 512 bf16 + 8 pad (16B) -> conflict-free fragment reads
#define LDSG_STRIDE 1284  // 1280 f32 + 4 pad

// ---------------------------------------------------------------------------
// Repack W (f32 [512,1280]) into bf16 WMMA B-fragments:
// layout [kb(16)][nt(80)][lane(32)][e(16)] so each lane does two b128 loads.
// ---------------------------------------------------------------------------
__global__ void repack_w(const float* __restrict__ W, __bf16* __restrict__ Wpk) {
    int flat = blockIdx.x * blockDim.x + threadIdx.x;
    if (flat >= KDIM * NDIM) return;
    int e    = flat & 15;
    int lane = (flat >> 4) & 31;
    int t3   = flat >> 9;
    int nt   = t3 % NTILES;
    int kb   = t3 / NTILES;
    int col    = nt * 16 + (lane & 15);
    int khalf  = lane >> 4;
    int klocal = (e < 8) ? (khalf * 8 + e) : (16 + khalf * 8 + (e - 8));
    int k      = kb * 32 + klocal;
    Wpk[flat] = (__bf16)W[k * NDIM + col];
}

// ---------------------------------------------------------------------------
// Leaf gather: leaves, leaves_aux, post-order leaf slots of internal_states,
// and bf16 level-0 h for the GEMM pipeline.
// ---------------------------------------------------------------------------
__global__ void gather_leaves(const int* __restrict__ tokens,
                              const float* __restrict__ emb,
                              const float* __restrict__ emb_aux,
                              float* __restrict__ leaves_out,
                              float* __restrict__ aux_out,
                              float* __restrict__ internal_out,
                              __bf16* __restrict__ h0) {
    int idx  = blockIdx.x * blockDim.x + threadIdx.x;   // < 128*512*256 = 2^24
    int h    = idx & (NHID - 1);
    int b    = (idx >> 8) & (BATCH - 1);
    int leaf = idx >> 17;                                // B*H = 2^17
    int tok  = tokens[leaf * BATCH + b];
    float v  = emb[(long)tok * NHID + h];
    float va = emb_aux[(long)tok * NHID + h];
    leaves_out[idx] = v;
    aux_out[idx]    = va;
    h0[idx]         = (__bf16)v;
    // post-order slot of leaf node (level 0, index `leaf`)
    int s = 0;
#pragma unroll
    for (int m = 0; m < 7; ++m) s += ((leaf >> m) & 1) * ((2 << m) - 1);
    internal_out[((long)s * BATCH + b) * NHID + h] = v;
}

__global__ void fill_ones(float* __restrict__ p, int n) {
    int i = blockIdx.x * blockDim.x + threadIdx.x;
    if (i < n) p[i] = 1.0f;
}

__device__ __forceinline__ float sigmoidf_(float x) {
    return 1.0f / (1.0f + __expf(-x));
}

// ---------------------------------------------------------------------------
// Fused level kernel: 16-row strip GEMM (bf16 WMMA, K=512, N=1280) + LSTM
// gates. 8 waves; wave w owns N-tiles [10w, 10w+10), split into two groups of
// 5 that are software-pipelined: B fragments for one group load from L2 while
// the other group's WMMAs execute. G staged in LDS so the 5H pre-activations
// never hit HBM.
// ---------------------------------------------------------------------------
__global__ __launch_bounds__(THREADS)
void tree_level(const __bf16* __restrict__ hin, const float* __restrict__ cin,
                const __bf16* __restrict__ Wpk, const float* __restrict__ bias,
                __bf16* __restrict__ hout, float* __restrict__ cout,
                float* __restrict__ internal_out, float* __restrict__ root_out,
                int level) {
    extern __shared__ char smem[];
    __bf16* As = (__bf16*)smem;                         // 16 x LDSA_STRIDE bf16
    float*  Gs = (float*)(smem + 16 * LDSA_STRIDE * 2); // 16 x LDSG_STRIDE f32

    const int t  = threadIdx.x;
    const int m0 = blockIdx.x * 16;        // global output row base
    const int p  = m0 >> 9;                // pair index (B = 512)
    const int b0 = m0 & (BATCH - 1);

    // ---- stage A strip: 16 rows x 512 k = (hL || hR), bf16, into LDS ----
    {
        int i   = t >> 4;                  // row 0..15
        int seg = t & 15;                  // 32 bf16 (64B) per segment
        int node = (seg < 8) ? (2 * p) : (2 * p + 1);
        int kc   = (seg & 7) * 32;
        const uint4* s4 = (const uint4*)(hin + ((long)node * BATCH + (b0 + i)) * NHID + kc);
        uint4* d4 = (uint4*)(As + i * LDSA_STRIDE + seg * 32);
        d4[0] = s4[0]; d4[1] = s4[1]; d4[2] = s4[2]; d4[3] = s4[3];
    }
    __syncthreads();

    const int wave  = t >> 5;
    const int lane  = t & 31;
    const int mfrag = lane & 15;
    const int khalf = lane >> 4;

    v8f acc[NT_PER_WAVE];
#pragma unroll
    for (int i = 0; i < NT_PER_WAVE; ++i) acc[i] = (v8f)(0.0f);

    auto loadA = [&](int kb) -> v16bf {
        v16bf a;
        const __bf16* abase = As + mfrag * LDSA_STRIDE + kb * 32 + khalf * 8;
        ((uint4*)&a)[0] = *(const uint4*)(abase);
        ((uint4*)&a)[1] = *(const uint4*)(abase + 16);
        return a;
    };
    // group g (0/1) = 5 consecutive N-tiles of this wave
    auto loadB = [&](v16bf* dst, int kb, int g) {
        const __bf16* wbase =
            Wpk + (((long)kb * NTILES + wave * NT_PER_WAVE + g * 5) * 32 + lane) * 16;
#pragma unroll
        for (int j = 0; j < 5; ++j) {
            const uint4* w4 = (const uint4*)(wbase + j * (32 * 16));
            ((uint4*)&dst[j])[0] = w4[0];
            ((uint4*)&dst[j])[1] = w4[1];
        }
    };

    // software pipeline: prime group0, then ping-pong groups across K steps
    v16bf bcur[5], bnext[5];
    v16bf a_cur = loadA(0);
    loadB(bcur, 0, 0);

#pragma unroll 1
    for (int kb = 0; kb < 16; ++kb) {
        loadB(bnext, kb, 1);                       // group1 of this kb
#pragma unroll
        for (int j = 0; j < 5; ++j)
            acc[j] = __builtin_amdgcn_wmma_f32_16x16x32_bf16(
                false, a_cur, false, bcur[j], (short)0, acc[j], false, false);

        v16bf a_next = a_cur;
        if (kb < 15) {
            a_next = loadA(kb + 1);
            loadB(bcur, kb + 1, 0);                // group0 of next kb
        }
#pragma unroll
        for (int j = 0; j < 5; ++j)
            acc[5 + j] = __builtin_amdgcn_wmma_f32_16x16x32_bf16(
                false, a_cur, false, bnext[j], (short)0, acc[5 + j], false, false);
        a_cur = a_next;
    }

    // ---- spill G tiles to LDS (C/D layout: lane<16 -> M=g, lane>=16 -> M=g+8)
    {
        const int mbase = (lane >> 4) * 8;
        const int ncol  = lane & 15;
#pragma unroll
        for (int nt = 0; nt < NT_PER_WAVE; ++nt) {
            int colbase = (wave * NT_PER_WAVE + nt) * 16 + ncol;
#pragma unroll
            for (int g = 0; g < 8; ++g)
                Gs[(mbase + g) * LDSG_STRIDE + colbase] = acc[nt][g];
        }
    }
    __syncthreads();

    // ---- LSTM gates: each thread handles one row x 16 hidden units ----
    {
        int row = t >> 4;
        int hb  = (t & 15) * 16;
        int b   = b0 + row;
        // post-order slot of node (level, p)
        int s = 0;
        for (int m = level; m <= 6; ++m)
            s += ((p >> (m - level)) & 1) * ((2 << m) - 1);
        int pos = s + (2 << level) - 2;

        const float* gr = Gs + row * LDSG_STRIDE;
        long cidxL = ((long)(2 * p) * BATCH + b) * NHID;
        long cidxR = ((long)(2 * p + 1) * BATCH + b) * NHID;
        long obase = ((long)p * BATCH + b) * NHID;
        long ibase = ((long)pos * BATCH + b) * NHID;

#pragma unroll 4
        for (int hh = hb; hh < hb + 16; ++hh) {
            float gi  = gr[hh]         + bias[hh];
            float gf1 = gr[256 + hh]   + bias[256 + hh];
            float gf2 = gr[512 + hh]   + bias[512 + hh];
            float go  = gr[768 + hh]   + bias[768 + hh];
            float gu  = gr[1024 + hh]  + bias[1024 + hh];
            float cL  = cin ? cin[cidxL + hh] : 0.0f;
            float cR  = cin ? cin[cidxR + hh] : 0.0f;
            float c = sigmoidf_(gi) * tanhf(gu)
                    + sigmoidf_(gf1) * cL
                    + sigmoidf_(gf2) * cR;
            float h = sigmoidf_(go) * tanhf(c);
            cout[obase + hh]         = c;
            hout[obase + hh]         = (__bf16)h;
            internal_out[ibase + hh] = h;
            if (root_out) root_out[(long)b * NHID + hh] = h;
        }
    }
}

// ---------------------------------------------------------------------------
extern "C" void kernel_launch(void* const* d_in, const int* in_sizes, int n_in,
                              void* d_out, int out_size, void* d_ws, size_t ws_size,
                              hipStream_t stream) {
    const int*   tokens = (const int*)d_in[0];
    const float* emb    = (const float*)d_in[1];
    const float* embax  = (const float*)d_in[2];
    const float* W      = (const float*)d_in[3];
    const float* bias   = (const float*)d_in[4];

    float* out = (float*)d_out;
    float* root_out     = out;                 //  512*256
    float* internal_out = out + 131072;        //  255*512*256
    float* imask        = out + 33554432;      //  255*512
    float* leaves_out   = out + 33684992;      //  128*512*256
    float* aux_out      = out + 50462208;      //  128*512*256
    float* lmask        = out + 67239424;      //  128*512

    char* ws = (char*)d_ws;
    size_t off = 0;
    auto wsalloc = [&](size_t bytes) -> char* {
        char* q = ws + off;
        off = (off + bytes + 255) & ~(size_t)255;
        return q;
    };
    __bf16* Wpk = (__bf16*)wsalloc((size_t)KDIM * NDIM * 2);
    __bf16* hA  = (__bf16*)wsalloc((size_t)128 * BATCH * NHID * 2);
    __bf16* hB  = (__bf16*)wsalloc((size_t)64  * BATCH * NHID * 2);
    float*  cA  = (float*) wsalloc((size_t)64  * BATCH * NHID * 4);
    float*  cB  = (float*) wsalloc((size_t)32  * BATCH * NHID * 4);

    repack_w<<<(KDIM * NDIM) / 256, 256, 0, stream>>>(W, Wpk);
    gather_leaves<<<(LEAVES * BATCH * NHID) / 256, 256, 0, stream>>>(
        tokens, emb, embax, leaves_out, aux_out, internal_out, hA);
    fill_ones<<<(130560 + 255) / 256, 256, 0, stream>>>(imask, 130560);
    fill_ones<<<(65536) / 256, 256, 0, stream>>>(lmask, 65536);

    const size_t lds_bytes = (size_t)16 * LDSA_STRIDE * 2   // A strip
                           + (size_t)16 * LDSG_STRIDE * 4;  // G strip (~97KB)

    __bf16* hcur  = hA;
    __bf16* hnext = hB;
    float*  ccur  = nullptr;
    for (int level = 1; level <= 7; ++level) {
        int P = LEAVES >> level;                 // pairs at this level
        float* cnext = (level & 1) ? cA : cB;
        float* root  = (level == 7) ? root_out : nullptr;
        int grid = P * (BATCH / 16);             // 16-row strips
        tree_level<<<grid, THREADS, lds_bytes, stream>>>(
            hcur, ccur, Wpk, bias, hnext, cnext, internal_out, root, level);
        __bf16* tmp = hcur; hcur = hnext; hnext = tmp;
        ccur = cnext;
    }
}
